// GraphAttentionEmbedding_42279658062343
// MI455X (gfx1250) — matrix operand order repaired and checked by reading, not compiled, not run on hardware
//
#include <hip/hip_runtime.h>
#include <hip/hip_bf16.h>
#include <math.h>

// ---------------------------------------------------------------------------
// TGN GraphAttentionEmbedding for MI455X (gfx1250, wave32, WMMA f16->f32)
// M-tiled GEMMs (32 rows/wave): each LDS B-fragment feeds 2 independent WMMAs.
// f16 intermediates (Q/K/V/Eproj) to cut HBM bytes.
// ---------------------------------------------------------------------------

typedef __attribute__((ext_vector_type(16))) _Float16 half16;
typedef __attribute__((ext_vector_type(8)))  _Float16 half8;
typedef __attribute__((ext_vector_type(8)))  float    float8;
typedef __attribute__((ext_vector_type(4)))  float    vfloat4;

#define N_NODES   100000
#define N_EDGES   500000
#define CH        128        // IN_CH == OUT_CH == MSG_DIM == TIME_DIM
#define HEADS     2
#define HC        64
#define NODE_WAVES (N_NODES/32)   // 3125  = 625 blocks * 5 waves (exact)
#define EDGE_WAVES (N_EDGES/32)   // 15625 = 3125 blocks * 5 waves (exact)
#define WAVES_PER_BLOCK 5
#define BLOCK_THREADS   160

// ---------------------------------------------------------------------------
// Fragment helpers (layouts per CDNA5 ISA §7.12.2, wave32)
// A 16x32 f16: lane<16 -> row M=lane, K = {0..7, 16..23}; lane>=16 -> K+8
// B 32x16 f16: lane<16 -> col N=lane, K = {0..7, 16..23}; lane>=16 -> K+8
// C/D 16x16 f32: VGPR r: lanes 0-15 -> (M=r, N=lane); lanes 16-31 -> (M=8+r)
// ---------------------------------------------------------------------------

// A fragment loaded from f32 source with in-register f16 conversion.
__device__ inline half16 load_a_frag_f32(const float* __restrict__ xf,
                                         int row, int lane, int kstep) {
  const int kbase = (lane < 16) ? 0 : 8;
  const float* p = xf + (size_t)row * CH + kstep * 32 + kbase;
  vfloat4 a0 = *(const vfloat4*)(p);
  vfloat4 a1 = *(const vfloat4*)(p + 4);
  vfloat4 b0 = *(const vfloat4*)(p + 16);
  vfloat4 b1 = *(const vfloat4*)(p + 20);
  half16 a;
#pragma unroll
  for (int i = 0; i < 4; ++i) {
    a[i]      = (_Float16)a0[i];
    a[4 + i]  = (_Float16)a1[i];
    a[8 + i]  = (_Float16)b0[i];
    a[12 + i] = (_Float16)b1[i];
  }
  return a;
}

__device__ inline half16 load_b_frag_lds(const _Float16* smem,
                                         int fragIdx, int lane) {
  const _Float16* p = smem + fragIdx * 512 + lane * 16;
  half8 lo = *(const half8*)(p);
  half8 hi = *(const half8*)(p + 8);
  half16 b;
#pragma unroll
  for (int i = 0; i < 8; ++i) { b[i] = lo[i]; b[i + 8] = hi[i]; }
  return b;
}

__device__ inline void store_tile_f16(const float8* acc, _Float16* __restrict__ dst,
                                      int row0, int lane) {
  int l15 = lane & 15;
#pragma unroll
  for (int nt = 0; nt < 8; ++nt) {
    int col = nt * 16 + l15;
#pragma unroll
    for (int r = 0; r < 8; ++r) {
      int row = row0 + ((lane < 16) ? r : 8 + r);
      dst[(size_t)row * CH + col] = (_Float16)acc[nt][r];
    }
  }
}

__device__ inline void store_tile_f32(const float8* acc, float* __restrict__ dst,
                                      int row0, int lane) {
  int l15 = lane & 15;
#pragma unroll
  for (int nt = 0; nt < 8; ++nt) {
    int col = nt * 16 + l15;
#pragma unroll
    for (int r = 0; r < 8; ++r) {
      int row = row0 + ((lane < 16) ? r : 8 + r);
      dst[(size_t)row * CH + col] = acc[nt][r];
    }
  }
}

// ---------------------------------------------------------------------------
// Prep kernels
// ---------------------------------------------------------------------------
// Repack W[k][n] (row-major, ld=128) into per-fragment lane layout:
// pack[(kt*8+nt)*512 + lane*16 + i] = W[kt*32 + ((lane<16)?0:8) + (i&7) + ((i>>3)<<4)][nt*16 + (lane&15)]
__global__ void __launch_bounds__(256) pack_w_kernel(
    const float* __restrict__ W, _Float16* __restrict__ pack, int kTiles) {
  int idx = blockIdx.x * 256 + threadIdx.x;
  int total = kTiles * 8 * 512;
  if (idx >= total) return;
  int frag = idx >> 9;
  int q    = idx & 511;
  int lane = q >> 4;
  int i    = q & 15;
  int kt   = frag >> 3;
  int nt   = frag & 7;
  int kloc = ((lane < 16) ? 0 : 8) + (i & 7) + ((i >> 3) << 4);
  int row  = kt * 32 + kloc;
  int col  = nt * 16 + (lane & 15);
  pack[idx] = (_Float16)W[(size_t)row * CH + col];
}

__global__ void __launch_bounds__(256) init_ms_kernel(
    float* __restrict__ mbuf, float* __restrict__ sbuf, int n) {
  int i = blockIdx.x * 256 + threadIdx.x;
  if (i < n) { mbuf[i] = -__builtin_inff(); sbuf[i] = 0.0f; }
}

// ---------------------------------------------------------------------------
// Node GEMM: Q/K/V = x@W+b -> ws (f16), skip = x@Wskip+bskip -> d_out (f32)
// One wave per 32-row tile (two 16-row WMMA tiles). A fragments (x, converted
// f32->f16 in-register) preloaded once and reused across all 4 matrices.
// B staged in LDS one matrix at a time (32 KB). All waves active -> uniform
// barriers, EXEC all-ones at every WMMA.
// ---------------------------------------------------------------------------
__global__ void __launch_bounds__(BLOCK_THREADS) node_gemm_kernel(
    const float* __restrict__ x,
    const _Float16* __restrict__ packQ, const _Float16* __restrict__ packK,
    const _Float16* __restrict__ packV, const _Float16* __restrict__ packS,
    const float* __restrict__ bq, const float* __restrict__ bk,
    const float* __restrict__ bv, const float* __restrict__ bs,
    _Float16* __restrict__ Q, _Float16* __restrict__ Kp,
    _Float16* __restrict__ V, float* __restrict__ out) {
  __shared__ _Float16 smem[16384];   // 32 KB: 32 fragments of 512 halfs

  int wave = blockIdx.x * WAVES_PER_BLOCK + (threadIdx.x >> 5);
  int lane = threadIdx.x & 31;
  int row0 = wave * 32;              // two 16-row tiles: row0, row0+16
  int l15  = lane & 15;

  const _Float16* packs[4]  = {packQ, packK, packV, packS};
  const float*    biases[4] = {bq, bk, bv, bs};

  // Preload A fragments for both tiles, all 4 k-steps (reused by 4 matrices).
  half16 a0[4], a1[4];
#pragma unroll
  for (int kk = 0; kk < 4; ++kk) {
    a0[kk] = load_a_frag_f32(x, row0 + l15,      lane, kk);
    a1[kk] = load_a_frag_f32(x, row0 + 16 + l15, lane, kk);
  }

#pragma unroll
  for (int mat = 0; mat < 4; ++mat) {
    __syncthreads();                 // previous mat's compute done
    for (int i = threadIdx.x; i < 2048; i += BLOCK_THREADS)
      ((half8*)smem)[i] = ((const half8*)packs[mat])[i];
    __syncthreads();

    float8 acc0[8], acc1[8];
#pragma unroll
    for (int nt = 0; nt < 8; ++nt) {
      float b = biases[mat][nt * 16 + l15];
#pragma unroll
      for (int r = 0; r < 8; ++r) { acc0[nt][r] = b; acc1[nt][r] = b; }
    }
#pragma unroll
    for (int kk = 0; kk < 4; ++kk) {
#pragma unroll
      for (int nt = 0; nt < 8; ++nt) {
        half16 bfr = load_b_frag_lds(smem, kk * 8 + nt, lane);
        acc0[nt] = __builtin_amdgcn_wmma_f32_16x16x32_f16(
            false, a0[kk], false, bfr, (short)0, acc0[nt], false, false);
        acc1[nt] = __builtin_amdgcn_wmma_f32_16x16x32_f16(
            false, a1[kk], false, bfr, (short)0, acc1[nt], false, false);
      }
    }
    if (mat == 0)      { store_tile_f16(acc0, Q,  row0, lane); store_tile_f16(acc1, Q,  row0 + 16, lane); }
    else if (mat == 1) { store_tile_f16(acc0, Kp, row0, lane); store_tile_f16(acc1, Kp, row0 + 16, lane); }
    else if (mat == 2) { store_tile_f16(acc0, V,  row0, lane); store_tile_f16(acc1, V,  row0 + 16, lane); }
    else               { store_tile_f32(acc0, out, row0, lane); store_tile_f32(acc1, out, row0 + 16, lane); }
  }
}

// ---------------------------------------------------------------------------
// Edge GEMM: e = [cos(rel_t*w+b) | msg] @ We -> Eproj (f16)
// One wave per 32-edge tile. A-frags built on the fly; K-blocks of 32 never
// straddle the 128 boundary, so the time-enc/msg branch is wave-uniform per
// k-step. We (64 KB packed) staged through a 32 KB LDS buffer in two halves.
// ---------------------------------------------------------------------------
__device__ inline half16 build_edge_a(float rel_t, int myrow, int kk, int kbase,
                                      const float* __restrict__ msg,
                                      const float* __restrict__ tw,
                                      const float* __restrict__ tb) {
  half16 a;
  int K0 = kk * 32 + kbase;
#pragma unroll
  for (int hs = 0; hs < 2; ++hs) {
    int Kb = K0 + hs * 16;
    if (Kb < CH) {               // time-encoding region
#pragma unroll
      for (int i = 0; i < 8; ++i) {
        int K = Kb + i;
        a[hs * 8 + i] = (_Float16)__builtin_cosf(rel_t * tw[K] + tb[K]);
      }
    } else {                      // msg region
      const float* mp = msg + (size_t)myrow * CH + (Kb - CH);
      vfloat4 m0 = *(const vfloat4*)(mp);
      vfloat4 m1 = *(const vfloat4*)(mp + 4);
#pragma unroll
      for (int i = 0; i < 4; ++i) {
        a[hs * 8 + i]     = (_Float16)m0[i];
        a[hs * 8 + 4 + i] = (_Float16)m1[i];
      }
    }
  }
  return a;
}

__global__ void __launch_bounds__(BLOCK_THREADS) edge_gemm_kernel(
    const float* __restrict__ t, const float* __restrict__ last_update,
    const long long* __restrict__ src_idx,
    const float* __restrict__ msg,
    const float* __restrict__ time_w, const float* __restrict__ time_b,
    const _Float16* __restrict__ packE,
    _Float16* __restrict__ Eproj) {
  __shared__ _Float16 smem[16384];   // 32 KB: 32 fragments

  int wave   = blockIdx.x * WAVES_PER_BLOCK + (threadIdx.x >> 5);
  int lane   = threadIdx.x & 31;
  int row0   = wave * 32;            // two 16-edge tiles
  int l15    = lane & 15;
  int myrow0 = row0 + l15;
  int myrow1 = row0 + 16 + l15;
  float rel_t0 = t[myrow0] - last_update[src_idx[myrow0]];
  float rel_t1 = t[myrow1] - last_update[src_idx[myrow1]];

  float8 acc0[8], acc1[8];
#pragma unroll
  for (int nt = 0; nt < 8; ++nt)
#pragma unroll
    for (int r = 0; r < 8; ++r) { acc0[nt][r] = 0.0f; acc1[nt][r] = 0.0f; }

  const int kbase = (lane < 16) ? 0 : 8;
#pragma unroll
  for (int stage = 0; stage < 2; ++stage) {
    __syncthreads();                 // previous stage's compute done
    for (int i = threadIdx.x; i < 2048; i += BLOCK_THREADS)
      ((half8*)smem)[i] = ((const half8*)packE)[stage * 2048 + i];
    __syncthreads();

#pragma unroll
    for (int kk4 = 0; kk4 < 4; ++kk4) {
      int kk = stage * 4 + kk4;
      half16 a0 = build_edge_a(rel_t0, myrow0, kk, kbase, msg, time_w, time_b);
      half16 a1 = build_edge_a(rel_t1, myrow1, kk, kbase, msg, time_w, time_b);
#pragma unroll
      for (int nt = 0; nt < 8; ++nt) {
        half16 bfr = load_b_frag_lds(smem, kk4 * 8 + nt, lane);
        acc0[nt] = __builtin_amdgcn_wmma_f32_16x16x32_f16(
            false, a0, false, bfr, (short)0, acc0[nt], false, false);
        acc1[nt] = __builtin_amdgcn_wmma_f32_16x16x32_f16(
            false, a1, false, bfr, (short)0, acc1[nt], false, false);
      }
    }
  }
  store_tile_f16(acc0, Eproj, row0, lane);
  store_tile_f16(acc1, Eproj, row0 + 16, lane);
}

// ---------------------------------------------------------------------------
// Segment softmax + scatter (f16 gathers, f32 math/atomics)
// ---------------------------------------------------------------------------
__device__ inline void atomicMaxFloat(float* addr, float val) {
  if (val >= 0.0f) atomicMax((int*)addr, __float_as_int(val));
  else             atomicMin((unsigned int*)addr, __float_as_uint(val));
}

__global__ void __launch_bounds__(256) alpha_dot_kernel(
    const long long* __restrict__ ei,
    const _Float16* __restrict__ Q, const _Float16* __restrict__ Kp,
    const _Float16* __restrict__ Eproj,
    float* __restrict__ alphaRaw, float* __restrict__ mbuf) {
  int tid = blockIdx.x * 256 + threadIdx.x;
  if (tid >= N_EDGES * HEADS) return;
  int e = tid >> 1, h = tid & 1;
  long long s = ei[e];
  long long d = ei[N_EDGES + e];
  const _Float16* q  = Q     + (size_t)d * CH + h * HC;
  const _Float16* k  = Kp    + (size_t)s * CH + h * HC;
  const _Float16* ep = Eproj + (size_t)e * CH + h * HC;
  float dot = 0.0f;
#pragma unroll
  for (int c = 0; c < HC; c += 8) {
    half8 qv = *(const half8*)(q + c);
    half8 kv = *(const half8*)(k + c);
    half8 ev = *(const half8*)(ep + c);
#pragma unroll
    for (int i = 0; i < 8; ++i)
      dot += (float)qv[i] * ((float)kv[i] + (float)ev[i]);
  }
  float val = dot * 0.125f;   // 1/sqrt(64)
  alphaRaw[tid] = val;
  atomicMaxFloat(&mbuf[(size_t)d * HEADS + h], val);
}

__global__ void __launch_bounds__(256) alpha_exp_kernel(
    const long long* __restrict__ ei, const float* __restrict__ mbuf,
    float* __restrict__ alpha, float* __restrict__ sbuf) {
  int tid = blockIdx.x * 256 + threadIdx.x;
  if (tid >= N_EDGES * HEADS) return;
  int e = tid >> 1, h = tid & 1;
  long long d = ei[N_EDGES + e];
  float a = __expf(alpha[tid] - mbuf[(size_t)d * HEADS + h]);
  alpha[tid] = a;
  atomicAdd(&sbuf[(size_t)d * HEADS + h], a);
}

__global__ void __launch_bounds__(256) scatter_kernel(
    const long long* __restrict__ ei,
    const _Float16* __restrict__ V, const _Float16* __restrict__ Eproj,
    const float* __restrict__ alpha, const float* __restrict__ sbuf,
    float* __restrict__ out) {
  int tid = blockIdx.x * 256 + threadIdx.x;   // E*128 = 64e6 < 2^31
  if (tid >= N_EDGES * CH) return;
  int e = tid >> 7;
  int c = tid & 127;
  int h = c >> 6;
  long long s = ei[e];
  long long d = ei[N_EDGES + e];
  float w = alpha[(size_t)e * HEADS + h] / (sbuf[(size_t)d * HEADS + h] + 1e-16f);
  float val = ((float)V[(size_t)s * CH + c] + (float)Eproj[(size_t)e * CH + c]) * w;
  atomicAdd(&out[(size_t)d * CH + c], val);
}

// ---------------------------------------------------------------------------
// Launcher
// ---------------------------------------------------------------------------
extern "C" void kernel_launch(void* const* d_in, const int* in_sizes, int n_in,
                              void* d_out, int out_size, void* d_ws, size_t ws_size,
                              hipStream_t stream) {
  (void)in_sizes; (void)n_in; (void)out_size; (void)ws_size;

  const float*     x           = (const float*)d_in[0];
  const float*     last_update = (const float*)d_in[1];
  const float*     t           = (const float*)d_in[2];
  const float*     msg         = (const float*)d_in[3];
  const long long* ei          = (const long long*)d_in[4];
  const float*     time_w      = (const float*)d_in[5];
  const float*     time_b      = (const float*)d_in[6];
  const float*     Wq          = (const float*)d_in[7];
  const float*     bq          = (const float*)d_in[8];
  const float*     Wk          = (const float*)d_in[9];
  const float*     bk          = (const float*)d_in[10];
  const float*     Wv          = (const float*)d_in[11];
  const float*     bv          = (const float*)d_in[12];
  const float*     We          = (const float*)d_in[13];
  const float*     Ws          = (const float*)d_in[14];
  const float*     bs          = (const float*)d_in[15];
  float* out = (float*)d_out;

  // -------- workspace carve-out (256B-aligned slices) --------
  char* ws = (char*)d_ws;
  size_t off = 0;
  auto carve = [&](size_t bytes) -> void* {
    void* p = ws + off;
    off += (bytes + 255) & ~(size_t)255;
    return p;
  };
  _Float16* packQ = (_Float16*)carve((size_t)4 * 8 * 512 * sizeof(_Float16));
  _Float16* packK = (_Float16*)carve((size_t)4 * 8 * 512 * sizeof(_Float16));
  _Float16* packV = (_Float16*)carve((size_t)4 * 8 * 512 * sizeof(_Float16));
  _Float16* packS = (_Float16*)carve((size_t)4 * 8 * 512 * sizeof(_Float16));
  _Float16* packE = (_Float16*)carve((size_t)8 * 8 * 512 * sizeof(_Float16));
  _Float16* Qb    = (_Float16*)carve((size_t)N_NODES * CH * sizeof(_Float16));
  _Float16* Kb    = (_Float16*)carve((size_t)N_NODES * CH * sizeof(_Float16));
  _Float16* Vb    = (_Float16*)carve((size_t)N_NODES * CH * sizeof(_Float16));
  _Float16* Eproj = (_Float16*)carve((size_t)N_EDGES * CH * sizeof(_Float16));
  float* alphaB   = (float*)carve((size_t)N_EDGES * HEADS * sizeof(float));
  float* mbuf     = (float*)carve((size_t)N_NODES * HEADS * sizeof(float));
  float* sbuf     = (float*)carve((size_t)N_NODES * HEADS * sizeof(float));

  // -------- prep --------
  pack_w_kernel<<<(4 * 8 * 512 + 255) / 256, 256, 0, stream>>>(Wq, packQ, 4);
  pack_w_kernel<<<(4 * 8 * 512 + 255) / 256, 256, 0, stream>>>(Wk, packK, 4);
  pack_w_kernel<<<(4 * 8 * 512 + 255) / 256, 256, 0, stream>>>(Wv, packV, 4);
  pack_w_kernel<<<(4 * 8 * 512 + 255) / 256, 256, 0, stream>>>(Ws, packS, 4);
  pack_w_kernel<<<(8 * 8 * 512 + 255) / 256, 256, 0, stream>>>(We, packE, 8);
  init_ms_kernel<<<(N_NODES * HEADS + 255) / 256, 256, 0, stream>>>(mbuf, sbuf, N_NODES * HEADS);

  // -------- GEMMs (WMMA) --------
  node_gemm_kernel<<<NODE_WAVES / WAVES_PER_BLOCK, BLOCK_THREADS, 0, stream>>>(
      x, packQ, packK, packV, packS, bq, bk, bv, bs, Qb, Kb, Vb, out);
  edge_gemm_kernel<<<EDGE_WAVES / WAVES_PER_BLOCK, BLOCK_THREADS, 0, stream>>>(
      t, last_update, ei, msg, time_w, time_b, packE, Eproj);

  // -------- segment softmax + scatter --------
  alpha_dot_kernel<<<(N_EDGES * HEADS + 255) / 256, 256, 0, stream>>>(
      ei, Qb, Kb, Eproj, alphaB, mbuf);
  alpha_exp_kernel<<<(N_EDGES * HEADS + 255) / 256, 256, 0, stream>>>(
      ei, mbuf, alphaB, sbuf);
  scatter_kernel<<<(N_EDGES * CH + 255) / 256, 256, 0, stream>>>(
      ei, Vb, Eproj, alphaB, sbuf, out);
}
